// HV_LCA_29592324669781
// MI455X (gfx1250) — matrix operand order repaired and checked by reading, not compiled, not run on hardware
//
#include <hip/hip_runtime.h>
#include <hip/hip_bf16.h>

// ---------------------------------------------------------------------------
// Shapes (match reference)
// ---------------------------------------------------------------------------
#define Pn     9216          // H*W = 96*96 sequence length / spatial positions
#define Wd     96
#define BB     2
#define NHEADS 4
#define NCHAIN 512           // B * HEADS * D_INNER = 2*4*64
#define NCHK   32            // scan chunks
#define CLEN   288           // 9216 / 32

// ---------------------------------------------------------------------------
// WMMA types (CDNA5, wave32): D(f32 16x16) = A(bf16 16x32) x B(bf16 32x16) + C
// ---------------------------------------------------------------------------
typedef __attribute__((ext_vector_type(16))) __bf16 v16bf;
typedef __attribute__((ext_vector_type(8)))  float  v8f;

union FragBF { v16bf v; unsigned short u[16]; uint4 q[2]; };

__device__ __forceinline__ unsigned short f2bf(float x) {
  unsigned u = __float_as_uint(x);
  u += 0x7fffu + ((u >> 16) & 1u);       // round-to-nearest-even
  return (unsigned short)(u >> 16);
}

// ---------------------------------------------------------------------------
// Generic GEMM:  C[m,n] = sum_k A[m,k] * B[k,n] (+ Add[m,n])
// A: fp32 weights [M,K] row-major. B: fp32 activations, row k at B + k*N.
// blockIdx.z enumerates (batch, head) via per-z pointer strides.
// Workgroup = 256 threads = 8 waves; tile 64(M) x 64(N); K stepped by 32.
// Each wave owns a 16x32 subtile -> two v_wmma_f32_16x16x32_bf16 per K-step.
//
// LDS tiles are stored in *fragment order* so each lane reads its whole
// 16-element bf16 fragment as two contiguous ds_load_b128:
//   A element (row r, k kk): half=(kk>>3)&1, e=(kk&7)+8*((kk>>4)&1)
//       -> As[r*40 + half*16 + e]           (row stride 80B, bank-staggered)
//   B element (k kk, col n):  half=(kk>>3)&1, e=(kk&7)+8*((kk>>4)&1)
//       -> Bs[n*40 + half*16 + e]
// Staging is branchless: float4 global loads with clamped indices + zero fill.
// Requires K % 4 == 0 and 16B-aligned row pitches (true for every call here).
// ---------------------------------------------------------------------------
__global__ __launch_bounds__(256) void gemm_wmma_bf16(
    const float* __restrict__ A, const float* __restrict__ Bm,
    const float* __restrict__ Add, float* __restrict__ C,
    int M, int K, int N, int heads,
    long Ah, long Bb, long Bh, long Cb, long Ch, long Db, long Dh)
{
  __shared__ __align__(16) unsigned short As[64 * 40];   // 5120 B
  __shared__ __align__(16) unsigned short Bs[64 * 40];   // 5120 B

  const int z  = blockIdx.z;
  const int hh = z % heads;
  const int bb = z / heads;
  A  += (long)hh * Ah;
  Bm += (long)bb * Bb + (long)hh * Bh;
  C  += (long)bb * Cb + (long)hh * Ch;
  if (Add) Add += (long)bb * Db + (long)hh * Dh;

  const int m0 = blockIdx.y * 64, n0 = blockIdx.x * 64;
  const int tid = threadIdx.x, lane = tid & 31, wave = tid >> 5;
  const int wm = wave & 3;          // 4 M-subtiles of 16
  const int wn = wave >> 2;         // 2 N-subtiles of 32
  const int l16 = lane & 15, half = lane >> 4;

  v8f c0 = {0,0,0,0,0,0,0,0};
  v8f c1 = {0,0,0,0,0,0,0,0};

  for (int k0 = 0; k0 < K; k0 += 32) {
    __syncthreads();
    // ---- stage A tile 64x32: 512 float4 loads, fragment-order b64 stores ----
    #pragma unroll
    for (int j = 0; j < 2; ++j) {
      int i  = tid + (j << 8);            // 0..511
      int r  = i >> 3;                    // tile row 0..63
      int c4 = (i & 7) << 2;              // k within tile: 0,4,...,28
      int gm = m0 + r;
      int gk = k0 + c4;
      bool in = (gm < M) && (gk < K);     // K%4==0 -> float4 fully in or out
      int  gmc = (gm < M) ? gm : (M - 1);
      long off = (gk < K) ? ((long)gmc * K + gk) : 0;
      float4 v = *reinterpret_cast<const float4*>(A + off);
      if (!in) { v.x = 0.f; v.y = 0.f; v.z = 0.f; v.w = 0.f; }
      uint2 pk;
      pk.x = (unsigned)f2bf(v.x) | ((unsigned)f2bf(v.y) << 16);
      pk.y = (unsigned)f2bf(v.z) | ((unsigned)f2bf(v.w) << 16);
      int hf = (c4 >> 3) & 1;
      int e0 = (c4 & 7) + (((c4 >> 4) & 1) << 3);
      *reinterpret_cast<uint2*>(&As[r * 40 + hf * 16 + e0]) = pk;
    }
    // ---- stage B tile 32x64: 512 float4 loads, fragment-order b16 stores ----
    #pragma unroll
    for (int j = 0; j < 2; ++j) {
      int i  = tid + (j << 8);
      int r  = i >> 4;                    // k within tile 0..31
      int c4 = (i & 15) << 2;             // n within tile: 0,4,...,60
      int gk = k0 + r;
      bool in = (gk < K);                 // N is a multiple of 64 for all calls
      long off = in ? ((long)gk * N + n0 + c4) : 0;
      float4 v = *reinterpret_cast<const float4*>(Bm + off);
      if (!in) { v.x = 0.f; v.y = 0.f; v.z = 0.f; v.w = 0.f; }
      int hf = (r >> 3) & 1, e = (r & 7) + (((r >> 4) & 1) << 3);
      int bu = c4 * 40 + hf * 16 + e;
      Bs[bu]       = f2bf(v.x);
      Bs[bu + 40]  = f2bf(v.y);
      Bs[bu + 80]  = f2bf(v.z);
      Bs[bu + 120] = f2bf(v.w);
    }
    if (k0 + 32 < K)   // gfx1250 global_prefetch of next B tile
      __builtin_prefetch(&Bm[(long)(k0 + 32) * N + n0 + (tid & 63)], 0, 1);
    __syncthreads();

    // ---- fragments: contiguous 32B per (row/col, half) -> 2x ds_load_b128 ----
    FragBF fa, fb0, fb1;
    {
      const uint4* ap = reinterpret_cast<const uint4*>(
          &As[(wm * 16 + l16) * 40 + half * 16]);
      fa.q[0] = ap[0]; fa.q[1] = ap[1];
      int nc = wn * 32 + l16;
      const uint4* bp0 = reinterpret_cast<const uint4*>(&Bs[nc * 40 + half * 16]);
      fb0.q[0] = bp0[0]; fb0.q[1] = bp0[1];
      const uint4* bp1 = reinterpret_cast<const uint4*>(&Bs[(nc + 16) * 40 + half * 16]);
      fb1.q[0] = bp1[0]; fb1.q[1] = bp1[1];
    }
    c0 = __builtin_amdgcn_wmma_f32_16x16x32_bf16(false, fa.v, false, fb0.v,
                                                 (short)0, c0, false, false);
    c1 = __builtin_amdgcn_wmma_f32_16x16x32_bf16(false, fa.v, false, fb1.v,
                                                 (short)0, c1, false, false);
  }

  // --- store C/D: VGPR r -> M = r + 8*half, lane -> N = l16 ---
  #pragma unroll
  for (int r = 0; r < 8; ++r) {
    int gm = m0 + wm * 16 + r + half * 8;
    int gn = n0 + wn * 32 + l16;
    if (gm < M) {
      float v0 = c0[r];
      float v1 = c1[r];
      if (Add) {
        v0 += Add[(long)gm * N + gn];
        v1 += Add[(long)gm * N + gn + 16];
      }
      C[(long)gm * N + gn]      = v0;
      C[(long)gm * N + gn + 16] = v1;
    }
  }
}

// ---------------------------------------------------------------------------
// LayerNorm over 128 channels, per (b, position). Loads coalesced in p.
// ---------------------------------------------------------------------------
__global__ __launch_bounds__(256) void ln_kernel(
    const float* __restrict__ in, const float* __restrict__ w,
    const float* __restrict__ b, float* __restrict__ out)
{
  long idx = (long)blockIdx.x * 256 + threadIdx.x;     // BB*Pn threads
  int  bb  = (int)(idx / Pn);
  long p   = idx % Pn;
  long base = (long)bb * 128 * Pn + p;
  float s = 0.f, s2 = 0.f;
  for (int c = 0; c < 128; ++c) {
    float v = in[base + (long)c * Pn];
    s += v; s2 = fmaf(v, v, s2);
  }
  float mu  = s * (1.f / 128.f);
  float var = s2 * (1.f / 128.f) - mu * mu;
  float inv = rsqrtf(var + 1e-5f);
  for (int c = 0; c < 128; ++c) {
    float v = in[base + (long)c * Pn];
    out[base + (long)c * Pn] = (v - mu) * inv * w[c] + b[c];
  }
}

// ---------------------------------------------------------------------------
// Depthwise 3x3, pad 1, over [B, Ctot, 96, 96]; processes channels [c0, c0+Cproc).
// mode 0: out = conv ; mode 1: out = tanh(conv) + in  (gated-residual branch)
// ---------------------------------------------------------------------------
__global__ __launch_bounds__(256) void dw3x3(
    const float* __restrict__ in, const float* __restrict__ w,
    float* __restrict__ out, int Cproc, int Ctot, int c0, int mode)
{
  int z = blockIdx.y;
  int c = z % Cproc, bb = z / Cproc;
  int p = blockIdx.x * 256 + threadIdx.x;
  if (p >= Pn) return;
  long base = ((long)bb * Ctot + c0 + c) * (long)Pn;
  int px = p % Wd, py = p / Wd;
  const float* wc = w + c * 9;
  float acc = 0.f;
  #pragma unroll
  for (int dy = -1; dy <= 1; ++dy) {
    int yy = py + dy; if (yy < 0 || yy >= Wd) continue;
    #pragma unroll
    for (int dx = -1; dx <= 1; ++dx) {
      int xx = px + dx; if (xx < 0 || xx >= Wd) continue;
      acc = fmaf(wc[(dy + 1) * 3 + (dx + 1)], in[base + yy * Wd + xx], acc);
    }
  }
  out[base + p] = (mode == 1) ? (tanhf(acc) + in[base + p]) : acc;
}

// fused = q_dw + k  (k = first 128 channels of 256-channel kv buffer)
__global__ void fuse_add(const float* __restrict__ q,
                         const float* __restrict__ kv, float* __restrict__ f)
{
  long i = (long)blockIdx.x * 256 + threadIdx.x;
  long per = 128L * Pn;
  if (i >= (long)BB * per) return;
  long b = i / per, r = i % per;
  f[i] = q[i] + kv[b * 256L * Pn + r];
}

// ---------------------------------------------------------------------------
// Causal depthwise conv1d (k=4, left pad 3) + SiLU along the sequence.
// Emits xc in GEMM layout [bh][d][l] AND transposed xcT [bh][l][d] for the scan.
// ---------------------------------------------------------------------------
__global__ __launch_bounds__(256) void cconv_silu(
    const float* __restrict__ xz, const float* __restrict__ cw,
    const float* __restrict__ cb, float* __restrict__ xc, float* __restrict__ xcT)
{
  int row = blockIdx.y;                      // 0..511 = bh*64 + d
  int bh = row >> 6, d = row & 63, h = bh & 3;
  int p = blockIdx.x * 256 + threadIdx.x;
  long xin = ((long)bh * 128 + d) * Pn;      // xi = first 64 rows of xz block
  const float* wv = cw + (((h << 6) + d) << 2);
  float acc = cb[(h << 6) + d];
  #pragma unroll
  for (int j = 0; j < 4; ++j) {
    int l = p - 3 + j;
    if (l >= 0) acc = fmaf(wv[j], xz[xin + l], acc);
  }
  float s = acc / (1.f + __expf(-acc));      // silu
  xc[(long)row * Pn + p] = s;
  xcT[((long)bh * Pn + p) * 64 + d] = s;
}

// dt = softplus(dbl[0:2] @ dt_w.T + dt_b), stored transposed [bh][l][d]
__global__ __launch_bounds__(256) void dt_softplus(
    const float* __restrict__ dbl, const float* __restrict__ dtw,
    const float* __restrict__ dtb, float* __restrict__ dtT)
{
  int row = blockIdx.y;
  int bh = row >> 6, d = row & 63, h = bh & 3;
  int p = blockIdx.x * 256 + threadIdx.x;
  long base = (long)bh * 34 * Pn;
  int  wi = ((h << 6) + d) << 1;
  float v = fmaf(dtw[wi], dbl[base + p],
            fmaf(dtw[wi + 1], dbl[base + Pn + p], dtb[(h << 6) + d]));
  float sp = (v > 20.f) ? v : log1pf(__expf(v));
  dtT[((long)bh * Pn + p) * 64 + d] = sp;
}

// ---------------------------------------------------------------------------
// Chunked selective scan: h_{l+1} = exp(dt*A)*h + dt*B*x (16 states in regs).
// Pass1: per (chain, chunk) local scan -> chunk-final product & state.
// Pass2: 32-step inter-chunk scan per (chain, s).
// Pass3: replay with carry-in; fuse y = sum_s h*C, + D*xc, * silu(z).
// chain = bh*64 + d is lane-fast -> dtT/xcT loads coalesced, B/C broadcast.
// ---------------------------------------------------------------------------
__global__ __launch_bounds__(256) void scan_pass1(
    const float* __restrict__ dtT, const float* __restrict__ xcT,
    const float* __restrict__ dbl, const float* __restrict__ A_log,
    float* __restrict__ Pend, float* __restrict__ Hend)
{
  int idx = blockIdx.x * 256 + threadIdx.x;       // 512*32
  int chain = idx & (NCHAIN - 1);
  int ci    = idx >> 9;
  int bh = chain >> 6, d = chain & 63, h = bh & 3;
  long tbase = (long)bh * Pn * 64 + d;
  long Bbase = ((long)bh * 34 + 2) * Pn;
  float Ar[16], hr[16], pa[16];
  #pragma unroll
  for (int s = 0; s < 16; ++s) {
    Ar[s] = -__expf(A_log[((h << 6) + d) * 16 + s]);
    hr[s] = 0.f; pa[s] = 1.f;
  }
  int l0 = ci * CLEN;
  for (int l = l0; l < l0 + CLEN; ++l) {
    float dtl = dtT[tbase + (long)l * 64];
    float xcl = xcT[tbase + (long)l * 64];
    #pragma unroll
    for (int s = 0; s < 16; ++s) {
      float dA = __expf(dtl * Ar[s]);
      float bv = dbl[Bbase + (long)s * Pn + l];   // uniform across wave
      hr[s] = fmaf(dA, hr[s], dtl * bv * xcl);
      pa[s] *= dA;
    }
  }
  long o = (long)chain * 16 * NCHK + ci;
  #pragma unroll
  for (int s = 0; s < 16; ++s) {
    Pend[o + (long)s * NCHK] = pa[s];
    Hend[o + (long)s * NCHK] = hr[s];
  }
}

__global__ void scan_pass2(const float* __restrict__ Pend,
                           const float* __restrict__ Hend,
                           float* __restrict__ Hin)
{
  int idx = blockIdx.x * 256 + threadIdx.x;       // 512*16
  if (idx >= NCHAIN * 16) return;
  long base = (long)idx * NCHK;
  float hc = 0.f;
  for (int ci = 0; ci < NCHK; ++ci) {
    Hin[base + ci] = hc;
    hc = fmaf(Pend[base + ci], hc, Hend[base + ci]);
  }
}

__global__ __launch_bounds__(256) void scan_pass3(
    const float* __restrict__ dtT, const float* __restrict__ xcT,
    const float* __restrict__ dbl, const float* __restrict__ A_log,
    const float* __restrict__ mD, const float* __restrict__ xz,
    const float* __restrict__ Hin, float* __restrict__ yt)
{
  int idx = blockIdx.x * 256 + threadIdx.x;
  int chain = idx & (NCHAIN - 1);
  int ci    = idx >> 9;
  int bh = chain >> 6, d = chain & 63, h = bh & 3;
  long tbase = (long)bh * Pn * 64 + d;
  long Bbase = ((long)bh * 34 + 2) * Pn;
  long Cbase = ((long)bh * 34 + 18) * Pn;
  long zbase = ((long)bh * 128 + 64 + d) * Pn;
  long ybase = (long)chain * Pn;
  float Dv = mD[(h << 6) + d];
  float Ar[16], hr[16];
  #pragma unroll
  for (int s = 0; s < 16; ++s) {
    Ar[s] = -__expf(A_log[((h << 6) + d) * 16 + s]);
    hr[s] = Hin[((long)chain * 16 + s) * NCHK + ci];
  }
  int l0 = ci * CLEN;
  for (int l = l0; l < l0 + CLEN; ++l) {
    float dtl = dtT[tbase + (long)l * 64];
    float xcl = xcT[tbase + (long)l * 64];
    float y = 0.f;
    #pragma unroll
    for (int s = 0; s < 16; ++s) {
      float dA = __expf(dtl * Ar[s]);
      float bv = dbl[Bbase + (long)s * Pn + l];
      hr[s] = fmaf(dA, hr[s], dtl * bv * xcl);
      y = fmaf(hr[s], dbl[Cbase + (long)s * Pn + l], y);
    }
    float zz  = xz[zbase + l];
    float sil = zz / (1.f + __expf(-zz));
    yt[ybase + l] = (y + Dv * xcl) * sil;         // GEMM layout [bh][d][l]
  }
}

// g = t1 * t2 after gated residuals (u holds both halves, 680-channel pitch)
__global__ void gate_mul(const float* __restrict__ u, float* __restrict__ g)
{
  long i = (long)blockIdx.x * 256 + threadIdx.x;
  long per = 340L * Pn;
  if (i >= (long)BB * per) return;
  long b = i / per, r = i % per;
  g[i] = u[b * 680L * Pn + r] * u[b * 680L * Pn + 340L * Pn + r];
}

// ---------------------------------------------------------------------------
// Orchestration
// ---------------------------------------------------------------------------
extern "C" void kernel_launch(void* const* d_in, const int* in_sizes, int n_in,
                              void* d_out, int out_size, void* d_ws, size_t ws_size,
                              hipStream_t stream) {
  (void)in_sizes; (void)n_in; (void)out_size; (void)ws_size;
  const float* x      = (const float*)d_in[0];
  const float* y      = (const float*)d_in[1];
  const float* ln_w   = (const float*)d_in[2];
  const float* ln_b   = (const float*)d_in[3];
  const float* q_w    = (const float*)d_in[4];
  const float* q_dw   = (const float*)d_in[5];
  const float* kv_w   = (const float*)d_in[6];
  const float* kv_dw  = (const float*)d_in[7];
  const float* o_w    = (const float*)d_in[8];
  const float* m_in_w = (const float*)d_in[9];
  const float* m_cw   = (const float*)d_in[10];
  const float* m_cb   = (const float*)d_in[11];
  const float* m_xp_w = (const float*)d_in[12];
  const float* m_dt_w = (const float*)d_in[13];
  const float* m_dt_b = (const float*)d_in[14];
  const float* m_Alog = (const float*)d_in[15];
  const float* m_D    = (const float*)d_in[16];
  const float* m_ow   = (const float*)d_in[17];
  const float* pi_w   = (const float*)d_in[18];
  const float* dw_w   = (const float*)d_in[19];
  const float* dw1_w  = (const float*)d_in[20];
  const float* dw2_w  = (const float*)d_in[21];
  const float* po_w   = (const float*)d_in[22];

  // workspace carve-up (~296 MB fp32 with aggressive buffer reuse)
  float* ws = (float*)d_ws;
  auto alloc = [&](size_t n) { float* p = ws; ws += n; return p; };
  const size_t mapB = (size_t)BB * 128 * Pn;
  float* xn   = alloc(mapB);
  float* yn   = alloc(mapB);
  float* q1   = alloc(mapB);
  float* qdw  = alloc(mapB);
  float* kv1  = alloc((size_t)BB * 256 * Pn);
  float* kvdw = alloc((size_t)BB * 256 * Pn);
  float* fus  = alloc(mapB);
  float* xz   = alloc((size_t)BB * NHEADS * 128 * Pn);
  float* xc   = alloc((size_t)BB * NHEADS * 64 * Pn);
  float* xcT  = alloc((size_t)BB * NHEADS * 64 * Pn);
  float* dblb = alloc((size_t)BB * NHEADS * 34 * Pn);
  float* dtT  = alloc((size_t)BB * NHEADS * 64 * Pn);
  float* Pend = alloc((size_t)NCHAIN * 16 * NCHK);
  float* Hend = alloc((size_t)NCHAIN * 16 * NCHK);
  float* Hin  = alloc((size_t)NCHAIN * 16 * NCHK);
  float* t    = alloc((size_t)BB * 680 * Pn);
  float* tdw  = alloc((size_t)BB * 680 * Pn);
  float* yt   = kv1;   // kv1 dead after kvdw
  float* mout = q1;    // q1 dead after qdw
  float* x2   = xn;    // xn dead after q gemm
  float* xg   = yn;    // yn dead after kv gemm
  float* u    = t;     // t dead after tdw
  float* g    = xz;    // xz dead after scan_pass3

  auto gemm = [&](const float* A, const float* Bm, const float* Add, float* C,
                  int M, int K, int N, int heads, int batch,
                  long Ah, long Bb_, long Bh_, long Cb_, long Ch_,
                  long Db_, long Dh_) {
    dim3 grid((N + 63) / 64, (M + 63) / 64, heads * batch);
    gemm_wmma_bf16<<<grid, dim3(256), 0, stream>>>(A, Bm, Add, C, M, K, N,
                                                   heads, Ah, Bb_, Bh_, Cb_,
                                                   Ch_, Db_, Dh_);
  };
  const long MP = 128L * Pn;

  // 1) LayerNorms
  ln_kernel<<<72, 256, 0, stream>>>(x, ln_w, ln_b, xn);
  ln_kernel<<<72, 256, 0, stream>>>(y, ln_w, ln_b, yn);
  // 2) q = dw3x3(1x1(xn))
  gemm(q_w, xn, nullptr, q1, 128, 128, Pn, 1, BB, 0, MP, 0, MP, 0, 0, 0);
  dw3x3<<<dim3(36, BB * 128), 256, 0, stream>>>(q1, q_dw, qdw, 128, 128, 0, 0);
  // 3) kv = dw3x3(1x1(yn))
  gemm(kv_w, yn, nullptr, kv1, 256, 128, Pn, 1, BB, 0, MP, 0, 256L * Pn, 0, 0, 0);
  dw3x3<<<dim3(36, BB * 256), 256, 0, stream>>>(kv1, kv_dw, kvdw, 256, 256, 0, 0);
  // 4) fused = q + k
  fuse_add<<<9216, 256, 0, stream>>>(qdw, kvdw, fus);
  // 5) per-head in-projection: xz[b,h,128,P] = in_w[h](128x32) x fused[b, h*32:, :]
  gemm(m_in_w, fus, nullptr, xz, 128, 32, Pn, NHEADS, BB,
       128L * 32, MP, 32L * Pn, 512L * Pn, MP, 0, 0);
  // 6) causal conv1d + silu -> xc (GEMM layout) and xcT (scan layout)
  cconv_silu<<<dim3(36, 512), 256, 0, stream>>>(xz, m_cw, m_cb, xc, xcT);
  // 7) x-projection: dbl[b,h,34,P] = xp_w[h](34x64) x xc
  gemm(m_xp_w, xc, nullptr, dblb, 34, 64, Pn, NHEADS, BB,
       34L * 64, 256L * Pn, 64L * Pn, 136L * Pn, 34L * Pn, 0, 0);
  // 8) dt = softplus(...) transposed for scan
  dt_softplus<<<dim3(36, 512), 256, 0, stream>>>(dblb, m_dt_w, m_dt_b, dtT);
  // 9) chunked selective scan (3 passes), fused D*xc + silu(z) gate
  scan_pass1<<<64, 256, 0, stream>>>(dtT, xcT, dblb, m_Alog, Pend, Hend);
  scan_pass2<<<32, 256, 0, stream>>>(Pend, Hend, Hin);
  scan_pass3<<<64, 256, 0, stream>>>(dtT, xcT, dblb, m_Alog, m_D, xz, Hin, yt);
  // 10) out-projection + v residual: mout[b, h*32+m, p]
  gemm(m_ow, yt, kvdw + 128L * Pn, mout, 32, 64, Pn, NHEADS, BB,
       32L * 64, 256L * Pn, 64L * Pn, MP, 32L * Pn, 256L * Pn, 32L * Pn);
  // 11) o_w 1x1 + residual x -> x2
  gemm(o_w, mout, x, x2, 128, 128, Pn, 1, BB, 0, MP, 0, MP, 0, MP, 0);
  // 12) final LN
  ln_kernel<<<72, 256, 0, stream>>>(x2, ln_w, ln_b, xg);
  // 13) pi 1x1 (128 -> 680), depthwise 3x3
  gemm(pi_w, xg, nullptr, t, 680, 128, Pn, 1, BB, 0, MP, 0, 680L * Pn, 0, 0, 0);
  dw3x3<<<dim3(36, BB * 680), 256, 0, stream>>>(t, dw_w, tdw, 680, 680, 0, 0);
  // 14) gated residual branches: u = tanh(dw(t_i)) + t_i
  dw3x3<<<dim3(36, BB * 340), 256, 0, stream>>>(tdw, dw1_w, u, 340, 680, 0, 1);
  dw3x3<<<dim3(36, BB * 340), 256, 0, stream>>>(tdw, dw2_w, u, 340, 680, 340, 1);
  // 15) g = t1 * t2 ; po 1x1 (340 -> 128) -> d_out
  gate_mul<<<24480, 256, 0, stream>>>(u, g);
  gemm(po_w, g, nullptr, (float*)d_out, 128, 340, Pn, 1, BB,
       0, 340L * Pn, 0, MP, 0, 0, 0);
}
